// HyperbolicTemporalEncoder_2637109920195
// MI455X (gfx1250) — matrix operand order repaired and checked by reading, not compiled, not run on hardware
//
#include <hip/hip_runtime.h>
#include <math.h>

// ---------------------------------------------------------------------------
// CDNA5 (gfx1250) wave32 WMMA types
// ---------------------------------------------------------------------------
typedef __attribute__((ext_vector_type(16))) __bf16 v16bf;
typedef __attribute__((ext_vector_type(8)))  float  v8f;

__device__ __forceinline__ float wave_sum(float x) {
#pragma unroll
  for (int o = 16; o > 0; o >>= 1) x += __shfl_xor(x, o, 32);
  return x;
}

// ---------------------------------------------------------------------------
// Utility: zero fill
// ---------------------------------------------------------------------------
__global__ void k_zero(float* p, long n) {
  long i = (long)blockIdx.x * blockDim.x + threadIdx.x;
  if (i < n) p[i] = 0.f;
}

// ---------------------------------------------------------------------------
// Edge time scatter: nt[s]+=t, nt[d]+=t, cnt[s]+=1, cnt[d]+=1, deg[d]+=1
// ---------------------------------------------------------------------------
__global__ void k_edge_time(const int* __restrict__ src, const int* __restrict__ dst,
                            const float* __restrict__ t,
                            float* nt, float* cnt, float* deg, int E) {
  int e = blockIdx.x * blockDim.x + threadIdx.x;
  if (e >= E) return;
  int s = src[e], d = dst[e];
  float tv = t[e];
  atomicAdd(&nt[s], tv);  atomicAdd(&nt[d], tv);
  atomicAdd(&cnt[s], 1.f); atomicAdd(&cnt[d], 1.f);
  atomicAdd(&deg[d], 1.f);
}

// ---------------------------------------------------------------------------
// x2 = x + cos(node_time * w_time + b_time)
// ---------------------------------------------------------------------------
__global__ void k_time_encode(const float* __restrict__ x,
                              const float* __restrict__ nt, const float* __restrict__ cnt,
                              const float* __restrict__ w_time, const float* __restrict__ b_time,
                              float* __restrict__ x2, int N) {
  int n = blockIdx.x;
  int d = threadIdx.x;           // 128 threads
  if (n >= N) return;
  float ntime = nt[n] / fmaxf(cnt[n], 1.f);
  x2[(long)n * 128 + d] = x[(long)n * 128 + d] + cosf(ntime * w_time[d] + b_time[d]);
}

// ---------------------------------------------------------------------------
// Feature scatter: agg[dst] += feat[src]  (wave per edge; 4 dims per lane)
// Node arrays are L2-resident (51 MB vs 192 MB L2) so atomics RMW in L2.
// ---------------------------------------------------------------------------
__global__ void k_edge_scatter(const int* __restrict__ src, const int* __restrict__ dst,
                               const float* __restrict__ feat, float* __restrict__ agg, int E) {
  int wid  = (blockIdx.x * blockDim.x + threadIdx.x) >> 5;
  int lane = threadIdx.x & 31;
  if (wid >= E) return;
  int s = src[wid], d = dst[wid];
  const float4 v = *(const float4*)(feat + (long)s * 128 + lane * 4);
  float* a = agg + (long)d * 128 + lane * 4;
  atomicAdd(a + 0, v.x); atomicAdd(a + 1, v.y);
  atomicAdd(a + 2, v.z); atomicAdd(a + 3, v.w);
}

// ---------------------------------------------------------------------------
// WMMA fragment helpers (native __bf16 casts -> v_cvt_pk_bf16_f32)
//   A layout (16x32, 16-bit): lanes 0-15 hold K 0..15 of row (lane&15),
//                             lanes 16-31 hold K 16..31.
//   B layout (32x16, 16-bit): lanes 0-15 hold K 0..15 of col (lane&15),
//                             lanes 16-31 hold K 16..31.
//   NOTE: out-of-range A rows are NOT zeroed -- row m of A only feeds row m of
//   D, and out-of-range D rows are masked at store. Pointers are just clamped.
// ---------------------------------------------------------------------------
__device__ __forceinline__ v16bf load_a_frag(const float* __restrict__ ap, int off,
                                             float rs, bool has_rs) {
  float4 p0 = *(const float4*)(ap + off);
  float4 p1 = *(const float4*)(ap + off + 4);
  float4 p2 = *(const float4*)(ap + off + 8);
  float4 p3 = *(const float4*)(ap + off + 12);
  const float v[16] = {p0.x, p0.y, p0.z, p0.w, p1.x, p1.y, p1.z, p1.w,
                       p2.x, p2.y, p2.z, p2.w, p3.x, p3.y, p3.z, p3.w};
  v16bf r;
  if (has_rs) {
#pragma unroll
    for (int j = 0; j < 16; ++j) r[j] = (__bf16)(v[j] * rs);
  } else {
#pragma unroll
    for (int j = 0; j < 16; ++j) r[j] = (__bf16)v[j];
  }
  return r;
}

__device__ __forceinline__ v16bf load_b_frag(const float* __restrict__ wp) {
  float4 q0 = *(const float4*)(wp + 0);
  float4 q1 = *(const float4*)(wp + 4);
  float4 q2 = *(const float4*)(wp + 8);
  float4 q3 = *(const float4*)(wp + 12);
  const float v[16] = {q0.x, q0.y, q0.z, q0.w, q1.x, q1.y, q1.z, q1.w,
                       q2.x, q2.y, q2.z, q2.w, q3.x, q3.y, q3.z, q3.w};
  v16bf r;
#pragma unroll
  for (int j = 0; j < 16; ++j) r[j] = (__bf16)v[j];
  return r;
}

__device__ __forceinline__ v16bf load_b_frag_t(const float* __restrict__ W,
                                               int krow, int Nout, int oc) {
  v16bf r;
#pragma unroll
  for (int j = 0; j < 16; ++j) r[j] = (__bf16)W[(long)(krow + j) * Nout + oc];
  return r;
}

// ---------------------------------------------------------------------------
// Generic GEMM: out[M,Nout] = act( (A*rowscale) @ W(^T) + bias + Cin )
//   A: [M,128] f32, W: [Nout,128] (or [128,Nout] if wtrans), K fixed at 128.
//   Block = 128 threads (4 waves). Each wave owns a 32x32 macro-tile of out
//   (2x2 of 16x16 WMMA tiles): per K-chunk 4 fragments -> 4 WMMAs.
//   Accumulators start at zero; Cin/bias/relu are applied in the epilogue so
//   the K-loop carries no extra live ranges (avoids scratch spills).
//   No early returns: EXEC stays all-ones for WMMA (ISA requirement).
// ---------------------------------------------------------------------------
__global__ void k_gemm(const float* __restrict__ A, const float* __restrict__ W,
                       const float* __restrict__ bias, const float* __restrict__ Cin,
                       const float* __restrict__ rowscale, float* __restrict__ out,
                       int M, int Nout, int relu, int wtrans) {
  const int lane = threadIdx.x & 31;
  const int wv   = threadIdx.x >> 5;
  const int row0 = blockIdx.x * 32;
  const int col0 = blockIdx.y * 128 + wv * 32;
  const int m    = lane & 15;
  const int half = lane >> 4;

  v8f c00 = {}, c01 = {}, c10 = {}, c11 = {};

  const int arow0 = min(row0 + m,      M - 1);   // clamped; no zeroing needed
  const int arow1 = min(row0 + 16 + m, M - 1);
  const bool has_rs = rowscale != nullptr;
  float rs0 = 1.f, rs1 = 1.f;
  if (has_rs) {
    rs0 = 1.f / fmaxf(rowscale[arow0], 1.f);
    rs1 = 1.f / fmaxf(rowscale[arow1], 1.f);
  }
  const float* ap0 = A + (long)arow0 * 128;
  const float* ap1 = A + (long)arow1 * 128;
  const int akh = half * 16;   // A half offset within K-chunk
  const int bkh = half * 16;   // B half offset within K-chunk
  const int oc0 = col0 + m, oc1 = col0 + 16 + m;

#pragma unroll
  for (int kc = 0; kc < 4; ++kc) {
    const int kb = kc * 32;
    v16bf a0 = load_a_frag(ap0, kb + akh, rs0, has_rs);
    v16bf a1 = load_a_frag(ap1, kb + akh, rs1, has_rs);
    v16bf b0, b1;
    if (!wtrans) {
      b0 = load_b_frag(W + (long)oc0 * 128 + kb + bkh);
      b1 = load_b_frag(W + (long)oc1 * 128 + kb + bkh);
    } else {
      b0 = load_b_frag_t(W, kb + bkh, Nout, oc0);
      b1 = load_b_frag_t(W, kb + bkh, Nout, oc1);
    }
    c00 = __builtin_amdgcn_wmma_f32_16x16x32_bf16(false, a0, false, b0, (short)0, c00, false, false);
    c01 = __builtin_amdgcn_wmma_f32_16x16x32_bf16(false, a0, false, b1, (short)0, c01, false, false);
    c10 = __builtin_amdgcn_wmma_f32_16x16x32_bf16(false, a1, false, b0, (short)0, c10, false, false);
    c11 = __builtin_amdgcn_wmma_f32_16x16x32_bf16(false, a1, false, b1, (short)0, c11, false, false);
  }

  // Epilogue: out = act(c + Cin + bias), guarded per output row.
  const float bias0 = bias ? bias[oc0] : 0.f;
  const float bias1 = bias ? bias[oc1] : 0.f;
#pragma unroll
  for (int r = 0; r < 8; ++r) {
    int rbase = row0 + r + half * 8;
#pragma unroll
    for (int ti = 0; ti < 2; ++ti) {
      int rr = rbase + 16 * ti;
      if (rr < M) {
        float v0 = ((ti == 0) ? c00[r] : c10[r]) + bias0;
        float v1 = ((ti == 0) ? c01[r] : c11[r]) + bias1;
        if (Cin) {
          v0 += Cin[(long)rr * Nout + oc0];
          v1 += Cin[(long)rr * Nout + oc1];
        }
        if (relu) { v0 = fmaxf(v0, 0.f); v1 = fmaxf(v1, 0.f); }
        out[(long)rr * Nout + oc0] = v0;
        out[(long)rr * Nout + oc1] = v1;
      }
    }
  }
}

// ---------------------------------------------------------------------------
// logmap0 per 128-dim row (wave per row); optionally records row validity
// (raw norm > 0; padded history slots are exactly zero by construction).
// ---------------------------------------------------------------------------
__global__ void k_logmap0(const float* __restrict__ y, float* __restrict__ out,
                          float* valid, int R) {
  int row  = blockIdx.x * 8 + (threadIdx.x >> 5);
  int lane = threadIdx.x & 31;
  if (row >= R) return;
  float4 v = *(const float4*)(y + (long)row * 128 + lane * 4);
  float ss  = v.x * v.x + v.y * v.y + v.z * v.z + v.w * v.w;
  float nrm = sqrtf(wave_sum(ss));
  if (valid && lane == 0) valid[row] = (nrm > 0.f) ? 1.f : 0.f;
  float sc1 = fminf((1.f - 1e-5f) / fmaxf(nrm, 1e-12f), 1.f);   // _project
  float np  = fmaxf(nrm * sc1, 1e-12f);
  float xc  = fminf(fmaxf(np, -1.f + 1e-6f), 1.f - 1e-6f);
  float at  = 0.5f * (log1pf(xc) - log1pf(-xc));                // artanh
  float s   = at * sc1 / np;
  float4 o  = make_float4(v.x * s, v.y * s, v.z * s, v.w * s);
  *(float4*)(out + (long)row * 128 + lane * 4) = o;
}

// ---------------------------------------------------------------------------
// Attention over W=3 history slots (wave per node).
// score_w = dot(q2, hist_tan_w)/sqrt(128) with q2 = Wk^T (Wq h_now);
// m = sum_w alpha_w * hist_tan_w   (ctx = m @ Wv^T done by k_gemm after).
// ---------------------------------------------------------------------------
__global__ void k_attn(const float* __restrict__ q2, const float* __restrict__ hist_tan,
                       const float* __restrict__ hvalid, float* __restrict__ mout, int N) {
  int n    = blockIdx.x * 8 + (threadIdx.x >> 5);
  int lane = threadIdx.x & 31;
  if (n >= N) return;
  float4 q = *(const float4*)(q2 + (long)n * 128 + lane * 4);
  const float inv = 0.08838834764831845f;  // 1/sqrt(128), TEMP=1
  float s[3], val[3];
#pragma unroll
  for (int w = 0; w < 3; ++w) {
    const float4 h = *(const float4*)(hist_tan + (long)n * 384 + w * 128 + lane * 4);
    float d = q.x * h.x + q.y * h.y + q.z * h.z + q.w * h.w;
    s[w]   = wave_sum(d) * inv;
    val[w] = hvalid[n * 3 + w];
  }
  float alpha[3] = {0.f, 0.f, 0.f};
  if (val[0] + val[1] + val[2] > 0.f) {
    float mx = -1e30f;
#pragma unroll
    for (int w = 0; w < 3; ++w) if (val[w] > 0.f) mx = fmaxf(mx, s[w]);
    float sum = 0.f;
#pragma unroll
    for (int w = 0; w < 3; ++w) { alpha[w] = (val[w] > 0.f) ? expf(s[w] - mx) : 0.f; sum += alpha[w]; }
#pragma unroll
    for (int w = 0; w < 3; ++w) alpha[w] /= sum;
  }
  float4 o = make_float4(0.f, 0.f, 0.f, 0.f);
#pragma unroll
  for (int w = 0; w < 3; ++w) {
    const float4 h = *(const float4*)(hist_tan + (long)n * 384 + w * 128 + lane * 4);
    o.x += alpha[w] * h.x; o.y += alpha[w] * h.y;
    o.z += alpha[w] * h.z; o.w += alpha[w] * h.w;
  }
  *(float4*)(mout + (long)n * 128 + lane * 4) = o;
}

// ---------------------------------------------------------------------------
// GRU combine + expmap0 + project + aux-loss accumulation (wave per node)
// ---------------------------------------------------------------------------
__global__ void k_final(const float* __restrict__ gi, const float* __restrict__ gh,
                        const float* __restrict__ hidden, const float* __restrict__ prev_hyp,
                        float* __restrict__ out, float* aux, int N) {
  int n    = blockIdx.x * 8 + (threadIdx.x >> 5);
  int lane = threadIdx.x & 31;
  if (n >= N) return;
  long b = (long)n * 384 + lane * 4;
  float ht[4]; float ss = 0.f;
#pragma unroll
  for (int i = 0; i < 4; ++i) {
    float r  = 1.f / (1.f + expf(-(gi[b + i] + gh[b + i])));
    float z  = 1.f / (1.f + expf(-(gi[b + 128 + i] + gh[b + 128 + i])));
    float nn = tanhf(gi[b + 256 + i] + r * gh[b + 256 + i]);
    float hd = hidden[(long)n * 128 + lane * 4 + i];
    float v  = (1.f - z) * nn + z * hd;
    ht[i] = v; ss += v * v;
  }
  float nrm = sqrtf(wave_sum(ss));
  float sc  = tanhf(nrm) / fmaxf(nrm, 1e-12f);                  // expmap0
  float pn  = nrm * sc;                                          // norm of expmap result
  float sc1 = fminf((1.f - 1e-5f) / fmaxf(pn, 1e-12f), 1.f);    // _project
  float xsq = 0.f, ysq = 0.f, dsq = 0.f, sab = 0.f;
#pragma unroll
  for (int i = 0; i < 4; ++i) {
    float e = ht[i] * sc * sc1;
    float p = prev_hyp[(long)n * 128 + lane * 4 + i];
    out[(long)n * 128 + lane * 4 + i] = e;
    xsq += e * e; ysq += p * p; dsq += (e - p) * (e - p); sab += fabsf(p);
  }
  xsq = wave_sum(xsq); ysq = wave_sum(ysq);
  dsq = wave_sum(dsq); sab = wave_sum(sab);
  if (lane == 0 && sab > 0.f) {
    float denom = fmaxf((1.f - xsq) * (1.f - ysq), 1e-8f);
    float za    = fmaxf(1.f + 2.f * dsq / denom, 1.f + 1e-6f);
    atomicAdd(&aux[0], acoshf(za));
    atomicAdd(&aux[1], 1.f);
  }
}

__global__ void k_aux(const float* aux, float* out, long pos) {
  out[pos] = 0.01f * aux[0] / fmaxf(aux[1], 1.f);   // 0 when no valid prev
}

// ---------------------------------------------------------------------------
// Orchestration
// ---------------------------------------------------------------------------
extern "C" void kernel_launch(void* const* d_in, const int* in_sizes, int n_in,
                              void* d_out, int out_size, void* d_ws, size_t ws_size,
                              hipStream_t stream) {
  const float* x        = (const float*)d_in[0];
  const int*   eidx     = (const int*)  d_in[1];
  const float* t        = (const float*)d_in[2];
  const float* prev_hyp = (const float*)d_in[3];
  const float* hist_hyp = (const float*)d_in[4];
  // d_in[5] = valid_mask: recomputed from zero-padded hist_hyp instead
  const float* w_time = (const float*)d_in[6];
  const float* b_time = (const float*)d_in[7];
  const float* W0_l   = (const float*)d_in[8];
  const float* b0_l   = (const float*)d_in[9];
  const float* W0_r   = (const float*)d_in[10];
  const float* W1_l   = (const float*)d_in[11];
  const float* b1_l   = (const float*)d_in[12];
  const float* W1_r   = (const float*)d_in[13];
  const float* W_ih   = (const float*)d_in[14];
  const float* W_hh   = (const float*)d_in[15];
  const float* b_ih   = (const float*)d_in[16];
  const float* b_hh   = (const float*)d_in[17];
  const float* Wq     = (const float*)d_in[18];
  const float* Wk     = (const float*)d_in[19];
  const float* Wv     = (const float*)d_in[20];

  const int  N    = in_sizes[0] / 128;
  const int  E    = in_sizes[2];
  const long slot = (long)N * 128;

  float* ws   = (float*)d_ws;
  float* nt   = ws;             // N
  float* cnt  = ws + N;         // N
  float* deg  = ws + 2L * N;    // N
  float* hval = ws + 3L * N;    // 3N
  float* aux  = ws + 6L * N;    // 2
  float* s1 = ws + 1 * slot;    // h           -> prev_tan
  float* s2 = ws + 2 * slot;    // agg         -> hidden
  float* s3 = ws + 3 * slot;    // x2 -> q -> m-> gh (first 128N floats)
  float* s4 = ws + 4 * slot;    // P  -> q2    -> gh (next 128N floats)
  float* s5 = ws + 5 * slot;    // h_now       -> gh (last 128N floats)
  float* s6 = ws + 6 * slot;    // hist_tan (3 slots) -> gi
  float* gh = s3;               // [N,384] spanning s3..s5
  float* gi = s6;               // [N,384] spanning s6..s8
  float* out = (float*)d_out;

  dim3 b128(128), b256(256);
  dim3 g128((N + 31) / 32, 1), g384((N + 31) / 32, 3);
  dim3 gN8((N + 7) / 8), gE((E + 255) / 256), gEw((E + 7) / 8);
  dim3 gslot((unsigned)((slot + 255) / 256));

  { long nz = 6L * N + 2;
    k_zero<<<dim3((unsigned)((nz + 255) / 256)), b256, 0, stream>>>(ws, nz); }
  k_zero<<<gslot, b256, 0, stream>>>(s2, slot);                                   // agg = 0
  k_edge_time<<<gE, b256, 0, stream>>>(eidx, eidx + E, t, nt, cnt, deg, E);
  k_time_encode<<<dim3(N), b128, 0, stream>>>(x, nt, cnt, w_time, b_time, s3, N); // x2
  k_edge_scatter<<<gEw, b256, 0, stream>>>(eidx, eidx + E, s3, s2, E);            // agg0
  k_gemm<<<g128, b128, 0, stream>>>(s2, W0_l, b0_l, nullptr, deg, s4, N, 128, 0, 0); // P
  k_gemm<<<g128, b128, 0, stream>>>(s3, W0_r, nullptr, s4, nullptr, s1, N, 128, 1, 0); // h=relu
  k_zero<<<gslot, b256, 0, stream>>>(s2, slot);                                   // agg = 0
  k_edge_scatter<<<gEw, b256, 0, stream>>>(eidx, eidx + E, s1, s2, E);            // agg1
  k_gemm<<<g128, b128, 0, stream>>>(s2, W1_l, b1_l, nullptr, deg, s4, N, 128, 0, 0); // P
  k_gemm<<<g128, b128, 0, stream>>>(s1, W1_r, nullptr, s4, nullptr, s5, N, 128, 0, 0); // h_now
  k_logmap0<<<gN8, b256, 0, stream>>>(prev_hyp, s1, nullptr, N);                  // prev_tan
  k_logmap0<<<dim3((3 * N + 7) / 8), b256, 0, stream>>>(hist_hyp, s6, hval, 3 * N); // hist_tan
  k_gemm<<<g128, b128, 0, stream>>>(s5, Wq, nullptr, nullptr, nullptr, s3, N, 128, 0, 0); // q
  k_gemm<<<g128, b128, 0, stream>>>(s3, Wk, nullptr, nullptr, nullptr, s4, N, 128, 0, 1); // q2=Wk^T q
  k_attn<<<gN8, b256, 0, stream>>>(s4, s6, hval, s3, N);                          // m
  k_gemm<<<g128, b128, 0, stream>>>(s3, Wv, nullptr, s1, nullptr, s2, N, 128, 0, 0); // hidden=m@Wv^T+prev_tan
  k_gemm<<<g384, b128, 0, stream>>>(s5, W_ih, b_ih, nullptr, nullptr, gi, N, 384, 0, 0); // gi
  k_gemm<<<g384, b128, 0, stream>>>(s2, W_hh, b_hh, nullptr, nullptr, gh, N, 384, 0, 0); // gh
  k_final<<<gN8, b256, 0, stream>>>(gi, gh, s2, prev_hyp, out, aux, N);
  k_aux<<<dim3(1), dim3(1), 0, stream>>>(aux, out, (long)N * 128);
}
